// DeepSeekMoELayer_69389491634336
// MI455X (gfx1250) — matrix operand order repaired and checked by reading, not compiled, not run on hardware
//
#include <hip/hip_runtime.h>

typedef __bf16 bf16;
typedef __attribute__((ext_vector_type(16))) __bf16 bf16x16;
typedef __attribute__((ext_vector_type(8)))  float  f32x8;

union AFrag { bf16x16 v; uint4 q[2]; };

// Problem dims (fixed by the reference)
constexpr int Tn = 2048;   // B*S tokens
constexpr int Hn = 1024;   // hidden
constexpr int FFn = 4096;  // shared intermediate
constexpr int En = 64;     // experts
constexpr int Kn = 8;      // top-k
constexpr int Mn = 512;    // moe intermediate

// Low 32 bits of a flat pointer into the LDS aperture are the LDS byte address
// (ISA 10.2).
static __device__ __forceinline__ unsigned ldsoff(const void* p) {
  return (unsigned)(unsigned long long)p;
}

// Async global->LDS copy of 16 bytes per lane, tracked by ASYNCcnt.
#define ASYNC_COPY_B128(lds_u32, gaddr_u64)                      \
  asm volatile("global_load_async_to_lds_b128 %0, %1, off"       \
               :: "v"(lds_u32), "v"(gaddr_u64) : "memory")

#define WAIT_ASYNC_4() asm volatile("s_wait_asynccnt 0x4" ::: "memory")
#define WAIT_ASYNC_0() asm volatile("s_wait_asynccnt 0x0" ::: "memory")

// ------------------------------------------------------------------
// f32 -> bf16 elementwise convert (n multiple of 4)
// ------------------------------------------------------------------
__global__ __launch_bounds__(256) void cvt_f32_bf16(const float* __restrict__ in,
                                                    bf16* __restrict__ out, int n) {
  int i = (blockIdx.x * 256 + threadIdx.x) * 4;
  if (i + 3 < n) {
    float4 v = *(const float4*)(in + i);
    out[i + 0] = (bf16)v.x;
    out[i + 1] = (bf16)v.y;
    out[i + 2] = (bf16)v.z;
    out[i + 3] = (bf16)v.w;
  }
}

// ------------------------------------------------------------------
// [R,C] f32 -> [C,R] bf16 (LDS-tiled transpose); R,C multiples of 64
// blockIdx.z batches experts via in/out strides
// ------------------------------------------------------------------
__global__ __launch_bounds__(256) void transpose_cvt(const float* __restrict__ in,
                                                     bf16* __restrict__ out,
                                                     int R, int C,
                                                     long inStride, long outStride) {
  __shared__ float tile[64][65];
  const float* ip = in + (size_t)blockIdx.z * inStride;
  bf16* op = out + (size_t)blockIdx.z * outStride;
  int r0 = blockIdx.y * 64, c0 = blockIdx.x * 64;
  int ci = threadIdx.x & 63, rb = threadIdx.x >> 6;
#pragma unroll
  for (int i = 0; i < 16; ++i) {
    int ri = rb + i * 4;
    tile[ri][ci] = ip[(size_t)(r0 + ri) * C + c0 + ci];
  }
  __syncthreads();
  int ri2 = threadIdx.x & 63, cb = threadIdx.x >> 6;
#pragma unroll
  for (int i = 0; i < 16; ++i) {
    int ci2 = cb + i * 4;
    op[(size_t)(c0 + ci2) * R + r0 + ri2] = (bf16)tile[ri2][ci2];
  }
}

// ------------------------------------------------------------------
// router: logits[t,e] = dot(x[t,:], Wg[:,e])   (f32, exact)
// ------------------------------------------------------------------
__global__ __launch_bounds__(256) void router_kernel(const float* __restrict__ x,
                                                     const float* __restrict__ Wg,
                                                     float* __restrict__ logits) {
  int t = blockIdx.x * 4 + (threadIdx.x >> 6);
  int e = threadIdx.x & 63;
  const float* xr = x + (size_t)t * Hn;
  float acc = 0.f;
  for (int k = 0; k < Hn; ++k) acc = fmaf(xr[k], Wg[(size_t)k * En + e], acc);
  logits[(size_t)t * En + e] = acc;
}

__global__ void zero_ints(int* p, int n) {
  int i = blockIdx.x * blockDim.x + threadIdx.x;
  if (i < n) p[i] = 0;
}

// ------------------------------------------------------------------
// per-token top-8 + softmax over selected; count tokens per expert
// ------------------------------------------------------------------
__global__ __launch_bounds__(256) void topk_kernel(const float* __restrict__ logits,
                                                   int* __restrict__ idx_out,
                                                   float* __restrict__ w_out,
                                                   int* __restrict__ counts) {
  int t = blockIdx.x * blockDim.x + threadIdx.x;
  if (t >= Tn) return;
  const float* lg = logits + (size_t)t * En;
  unsigned long long mask = 0ull;
  float vals[Kn];
  int ids[Kn];
#pragma unroll
  for (int kk = 0; kk < Kn; ++kk) {
    float best = -3.0e38f;
    int bi = 0;
    for (int j = 0; j < En; ++j) {
      if ((mask >> j) & 1ull) continue;
      float lv = lg[j];
      if (lv > best) { best = lv; bi = j; }
    }
    mask |= (1ull << bi);
    vals[kk] = best;
    ids[kk] = bi;
  }
  float m = vals[0], s = 0.f;
#pragma unroll
  for (int kk = 0; kk < Kn; ++kk) { vals[kk] = __expf(vals[kk] - m); s += vals[kk]; }
  float inv = 1.f / s;
#pragma unroll
  for (int kk = 0; kk < Kn; ++kk) {
    idx_out[t * Kn + kk] = ids[kk];
    w_out[t * Kn + kk] = vals[kk] * inv;
    atomicAdd(&counts[ids[kk]], 1);
  }
}

__global__ void scan_offsets(const int* __restrict__ counts, int* __restrict__ offsets) {
  if (threadIdx.x == 0 && blockIdx.x == 0) {
    int s = 0;
    for (int e = 0; e < En; ++e) { offsets[e] = s; s += counts[e]; }
  }
}

__global__ __launch_bounds__(256) void fill_lists(const int* __restrict__ topk_idx,
                                                  const float* __restrict__ topk_w,
                                                  const int* __restrict__ offsets,
                                                  int* __restrict__ cursor,
                                                  int* __restrict__ list_tok,
                                                  float* __restrict__ list_w) {
  int i = blockIdx.x * blockDim.x + threadIdx.x;
  if (i >= Tn * Kn) return;
  int e = topk_idx[i];
  int pos = atomicAdd(&cursor[e], 1);
  int dst = offsets[e] + pos;
  list_tok[dst] = i / Kn;
  list_w[dst] = topk_w[i];
}

// ------------------------------------------------------------------
// bf16 NT-GEMM mainloop: C[m,n] += sum_k A[m,k]*B[n,k]
// block tile 128x128x32, 256 threads (8 waves), wave tile 64x32 (4x2 wmma).
// Tiles staged global->LDS with async copies (ASYNCcnt), double-buffered;
// pipeline unrolled x2 so the buffer index is compile-time constant and the
// steady-state body is branch-free (K/32 is even for every GEMM here).
// A rows optionally gathered (indices in `gather`), clamped to cnt-1.
// ------------------------------------------------------------------
__device__ __forceinline__ void gemm_mainloop(const bf16* __restrict__ A, int lda,
                                              const bf16* __restrict__ B, int ldb,
                                              int K, int m0, int n0,
                                              const int* __restrict__ gather, int cnt,
                                              f32x8 acc[4][2]) {
  // padded rows: 80B = 5x16B keeps every 16B chunk aligned for b128
  __shared__ alignas(16) bf16 As[2][128][40];
  __shared__ alignas(16) bf16 Bs[2][128][40];
  const int tid = threadIdx.x;
  const int lane = tid & 31;
  const int w = tid >> 5;
  const int wm = w & 1;        // 0..1 : 64-row half
  const int wn = w >> 1;       // 0..3 : 32-col quarter
  const int lrow = tid >> 1;   // 0..127 staging row
  const int kseg = (tid & 1) * 16;

  int arow = m0 + lrow;
  if (arow >= cnt) arow = cnt - 1;
  int asrc = gather ? gather[arow] : arow;
  const bf16* aptr = A + (size_t)asrc * lda + kseg;
  const bf16* bptr = B + (size_t)(n0 + lrow) * ldb + kseg;

  unsigned la0[2], la1[2], lb0[2], lb1[2];
#pragma unroll
  for (int s = 0; s < 2; ++s) {
    la0[s] = ldsoff(&As[s][lrow][kseg]);
    la1[s] = ldsoff(&As[s][lrow][kseg + 8]);
    lb0[s] = ldsoff(&Bs[s][lrow][kseg]);
    lb1[s] = ldsoff(&Bs[s][lrow][kseg + 8]);
  }

  const int r = lane & 15;
  const int kh = lane >> 4;
  const int niter = K >> 5;  // even (16 / 32 / 128)

  // issue async copies for the next K-tile into buffer s
  auto issue = [&](int s) {
    ASYNC_COPY_B128(la0[s], (unsigned long long)aptr);
    ASYNC_COPY_B128(la1[s], (unsigned long long)(aptr + 8));
    ASYNC_COPY_B128(lb0[s], (unsigned long long)bptr);
    ASYNC_COPY_B128(lb1[s], (unsigned long long)(bptr + 8));
    aptr += 32;
    bptr += 32;
  };

  // 6 fragments (12 b128 ds_loads) + 8 WMMAs from buffer s
  auto compute = [&](int s) {
    AFrag af[4], bfr[2];
#pragma unroll
    for (int ti = 0; ti < 4; ++ti) {
      int ar = wm * 64 + ti * 16 + r;
      af[ti].q[0] = *(const uint4*)&As[s][ar][kh * 8];       // K 0-7 / 8-15
      af[ti].q[1] = *(const uint4*)&As[s][ar][16 + kh * 8];  // K 16-23 / 24-31
    }
#pragma unroll
    for (int tj = 0; tj < 2; ++tj) {
      int br = wn * 32 + tj * 16 + r;
      bfr[tj].q[0] = *(const uint4*)&Bs[s][br][kh * 8];       // K 0-7 / 8-15
      bfr[tj].q[1] = *(const uint4*)&Bs[s][br][16 + kh * 8];  // K 16-23 / 24-31
    }
#pragma unroll
    for (int ti = 0; ti < 4; ++ti)
#pragma unroll
      for (int tj = 0; tj < 2; ++tj)
        acc[ti][tj] = __builtin_amdgcn_wmma_f32_16x16x32_bf16(
            false, af[ti].v, false, bfr[tj].v, (short)0, acc[ti][tj], false, false);
  };

  // prologue: tile 0 -> buffer 0
  issue(0);

  // steady state: branch-free, two tiles per iteration, constant buffer ids
  for (int i = 0; i + 2 < niter; i += 2) {
    issue(1);          // tile i+1
    WAIT_ASYNC_4();    // tile i's copies complete (4 newer still in flight)
    __syncthreads();   // all waves' copies for buffer 0 landed
    compute(0);        // tile i
    __syncthreads();   // buffer 0 free for refill
    issue(0);          // tile i+2
    WAIT_ASYNC_4();
    __syncthreads();
    compute(1);        // tile i+1
    __syncthreads();
  }

  // tail: tiles niter-2 and niter-1
  issue(1);
  WAIT_ASYNC_4();
  __syncthreads();
  compute(0);
  __syncthreads();
  WAIT_ASYNC_0();
  __syncthreads();
  compute(1);
}

// epilogue A: SiLU -> bf16 store (shared up-proj / expert up-proj w/ gather)
__global__ __launch_bounds__(256) void gemm_silu_bf16(const bf16* __restrict__ A,
                                                      const bf16* __restrict__ B,
                                                      bf16* __restrict__ C,
                                                      int K, int lda, int ldb, int ldc,
                                                      long Bstride, int Mrows,
                                                      const int* __restrict__ list_tok,
                                                      const int* __restrict__ offsets,
                                                      const int* __restrict__ counts) {
  int e = blockIdx.z;
  int m0 = blockIdx.y * 128, n0 = blockIdx.x * 128;
  int seg = 0, cnt = Mrows;
  const int* gather = nullptr;
  if (counts) {
    seg = offsets[e];
    cnt = counts[e];
    if (m0 >= cnt) return;  // uniform across block: EXEC stays all-1 for WMMA
    gather = list_tok + seg;
  }
  const bf16* Bp = B + (size_t)e * Bstride;
  bf16* Cp = C + (size_t)seg * ldc;

  f32x8 z = {0.f, 0.f, 0.f, 0.f, 0.f, 0.f, 0.f, 0.f};
  f32x8 acc[4][2];
#pragma unroll
  for (int i = 0; i < 4; ++i)
#pragma unroll
    for (int j = 0; j < 2; ++j) acc[i][j] = z;

  gemm_mainloop(A, lda, Bp, ldb, K, m0, n0, gather, cnt, acc);

  int lane = threadIdx.x & 31, w = threadIdx.x >> 5;
  int wm = w & 1, wn = w >> 1;
  int nn = lane & 15, kh = lane >> 4;
#pragma unroll
  for (int ti = 0; ti < 4; ++ti)
#pragma unroll
    for (int v = 0; v < 8; ++v) {
      int gm = m0 + wm * 64 + ti * 16 + v + kh * 8;
      if (gm < cnt) {
#pragma unroll
        for (int tj = 0; tj < 2; ++tj) {
          int gn = n0 + wn * 32 + tj * 16 + nn;
          float xv = acc[ti][tj][v];
          float sv = xv / (1.f + __expf(-xv));  // silu
          Cp[(size_t)gm * ldc + gn] = (bf16)sv;
        }
      }
    }
}

// epilogue B: plain f32 store (shared down-proj; writes whole output)
__global__ __launch_bounds__(256) void gemm_f32_store(const bf16* __restrict__ A,
                                                      const bf16* __restrict__ B,
                                                      float* __restrict__ C,
                                                      int K, int lda, int ldb, int ldc,
                                                      int Mrows) {
  int m0 = blockIdx.y * 128, n0 = blockIdx.x * 128;
  f32x8 z = {0.f, 0.f, 0.f, 0.f, 0.f, 0.f, 0.f, 0.f};
  f32x8 acc[4][2];
#pragma unroll
  for (int i = 0; i < 4; ++i)
#pragma unroll
    for (int j = 0; j < 2; ++j) acc[i][j] = z;

  gemm_mainloop(A, lda, B, ldb, K, m0, n0, nullptr, Mrows, acc);

  int lane = threadIdx.x & 31, w = threadIdx.x >> 5;
  int wm = w & 1, wn = w >> 1;
  int nn = lane & 15, kh = lane >> 4;
#pragma unroll
  for (int ti = 0; ti < 4; ++ti)
#pragma unroll
    for (int v = 0; v < 8; ++v) {
      int gm = m0 + wm * 64 + ti * 16 + v + kh * 8;
#pragma unroll
      for (int tj = 0; tj < 2; ++tj) {
        int gn = n0 + wn * 32 + tj * 16 + nn;
        C[(size_t)gm * ldc + gn] = acc[ti][tj][v];
      }
    }
}

// epilogue C: weighted scatter-add into output (expert down-proj)
__global__ __launch_bounds__(256) void gemm_scatter_add(const bf16* __restrict__ A,
                                                        const bf16* __restrict__ B,
                                                        float* __restrict__ out,
                                                        int K, int lda, int ldb, int ldc,
                                                        long Bstride,
                                                        const int* __restrict__ list_tok,
                                                        const float* __restrict__ list_w,
                                                        const int* __restrict__ offsets,
                                                        const int* __restrict__ counts) {
  int e = blockIdx.z;
  int m0 = blockIdx.y * 128, n0 = blockIdx.x * 128;
  int seg = offsets[e], cnt = counts[e];
  if (m0 >= cnt) return;  // uniform exit
  const bf16* Ap = A + (size_t)seg * lda;
  const bf16* Bp = B + (size_t)e * Bstride;

  f32x8 z = {0.f, 0.f, 0.f, 0.f, 0.f, 0.f, 0.f, 0.f};
  f32x8 acc[4][2];
#pragma unroll
  for (int i = 0; i < 4; ++i)
#pragma unroll
    for (int j = 0; j < 2; ++j) acc[i][j] = z;

  gemm_mainloop(Ap, lda, Bp, ldb, K, m0, n0, nullptr, cnt, acc);

  int lane = threadIdx.x & 31, w = threadIdx.x >> 5;
  int wm = w & 1, wn = w >> 1;
  int nn = lane & 15, kh = lane >> 4;
#pragma unroll
  for (int ti = 0; ti < 4; ++ti)
#pragma unroll
    for (int v = 0; v < 8; ++v) {
      int lr = m0 + wm * 64 + ti * 16 + v + kh * 8;
      if (lr < cnt) {
        int t = list_tok[seg + lr];
        float wt = list_w[seg + lr];
#pragma unroll
        for (int tj = 0; tj < 2; ++tj) {
          int gn = n0 + wn * 32 + tj * 16 + nn;
          atomicAdd(&out[(size_t)t * ldc + gn], wt * acc[ti][tj][v]);
        }
      }
    }
}

// ------------------------------------------------------------------
extern "C" void kernel_launch(void* const* d_in, const int* in_sizes, int n_in,
                              void* d_out, int out_size, void* d_ws, size_t ws_size,
                              hipStream_t stream) {
  const float* x   = (const float*)d_in[0];  // [T,H]
  const float* Ws1 = (const float*)d_in[1];  // [H,FF]
  const float* Ws2 = (const float*)d_in[2];  // [FF,H]
  const float* Wg  = (const float*)d_in[3];  // [H,E]
  const float* W1  = (const float*)d_in[4];  // [E,H,M]
  const float* W2  = (const float*)d_in[5];  // [E,M,H]

  float* out = (float*)d_out;                        // [T,H]
  float* logits = out + (size_t)Tn * Hn;             // [T,E]

  // workspace layout
  char* ws = (char*)d_ws;
  size_t o = 0;
  bf16* xb   = (bf16*)(ws + o); o += (size_t)Tn * Hn * 2;
  bf16* Ws1t = (bf16*)(ws + o); o += (size_t)FFn * Hn * 2;     // [FF,H]
  bf16* Ws2t = (bf16*)(ws + o); o += (size_t)Hn * FFn * 2;     // [H,FF]
  bf16* W1t  = (bf16*)(ws + o); o += (size_t)En * Mn * Hn * 2; // [E][M,H]
  bf16* W2t  = (bf16*)(ws + o); o += (size_t)En * Hn * Mn * 2; // [E][H,M]
  bf16* h1b  = (bf16*)(ws + o); o += (size_t)Tn * FFn * 2;     // shared act
  bf16* moeh = (bf16*)(ws + o); o += (size_t)Tn * Kn * Mn * 2; // expert act segments
  int*   tidx    = (int*)(ws + o);   o += (size_t)Tn * Kn * 4;
  float* tw      = (float*)(ws + o); o += (size_t)Tn * Kn * 4;
  int*   counts  = (int*)(ws + o);   o += En * 4;
  int*   cursor  = (int*)(ws + o);   o += En * 4;  // contiguous with counts
  int*   offsets = (int*)(ws + o);   o += En * 4;
  int*   ltok    = (int*)(ws + o);   o += (size_t)Tn * Kn * 4;
  float* lw      = (float*)(ws + o); o += (size_t)Tn * Kn * 4;
  if (o > ws_size) return;  // insufficient scratch (should not happen)

  // 1) activations -> bf16
  cvt_f32_bf16<<<(Tn * Hn) / 1024, 256, 0, stream>>>(x, xb, Tn * Hn);

  // 2) weights -> bf16, transposed to [N,K]
  transpose_cvt<<<dim3(FFn / 64, Hn / 64, 1), 256, 0, stream>>>(Ws1, Ws1t, Hn, FFn, 0, 0);
  transpose_cvt<<<dim3(Hn / 64, FFn / 64, 1), 256, 0, stream>>>(Ws2, Ws2t, FFn, Hn, 0, 0);
  transpose_cvt<<<dim3(Mn / 64, Hn / 64, En), 256, 0, stream>>>(
      W1, W1t, Hn, Mn, (long)Hn * Mn, (long)Mn * Hn);
  transpose_cvt<<<dim3(Hn / 64, Mn / 64, En), 256, 0, stream>>>(
      W2, W2t, Mn, Hn, (long)Mn * Hn, (long)Hn * Mn);

  // 3) router logits (f32, written straight into d_out's logits region)
  router_kernel<<<Tn / 4, 256, 0, stream>>>(x, Wg, logits);

  // 4) top-k + softmax + expert lists
  zero_ints<<<1, 128, 0, stream>>>(counts, 2 * En);  // counts + cursor
  topk_kernel<<<Tn / 256, 256, 0, stream>>>(logits, tidx, tw, counts);
  scan_offsets<<<1, 32, 0, stream>>>(counts, offsets);
  fill_lists<<<(Tn * Kn) / 256, 256, 0, stream>>>(tidx, tw, offsets, cursor, ltok, lw);

  // 5) shared expert: h1 = silu(x @ Ws1) ; out = h1 @ Ws2 (plain store, first writer)
  gemm_silu_bf16<<<dim3(FFn / 128, Tn / 128, 1), 256, 0, stream>>>(
      xb, Ws1t, h1b, Hn, Hn, Hn, FFn, 0, Tn, nullptr, nullptr, nullptr);
  gemm_f32_store<<<dim3(Hn / 128, Tn / 128, 1), 256, 0, stream>>>(
      h1b, Ws2t, out, FFn, FFn, FFn, Hn, Tn);

  // 6) experts (gathered): moeh = silu(x[gather] @ W1[e]) ; out += w * (moeh @ W2[e])
  gemm_silu_bf16<<<dim3(Mn / 128, Tn / 128, En), 256, 0, stream>>>(
      xb, W1t, moeh, Hn, Hn, Hn, Mn, (long)Mn * Hn, Tn, ltok, offsets, counts);
  gemm_scatter_add<<<dim3(Hn / 128, Tn / 128, En), 256, 0, stream>>>(
      moeh, W2t, out, Mn, Mn, Mn, Hn, (long)Hn * Mn, ltok, lw, offsets, counts);
}
